// MultiHeadAttention_335007449165
// MI455X (gfx1250) — matrix operand order repaired and checked by reading, not compiled, not run on hardware
//
#include <hip/hip_runtime.h>

// Problem constants (match reference)
#define BB   4
#define SS   2048
#define QSZ  512
#define KSZ  1024
#define FF   1024
#define HH   16
#define DHH  64
#define MTOT (BB * SS)   // 8192

typedef __attribute__((ext_vector_type(16))) __bf16 v16bf;
typedef __attribute__((ext_vector_type(8)))  __bf16 v8bf;
typedef __attribute__((ext_vector_type(8)))  float  v8f;

union FragA {
    v16bf v;
    v8bf  h[2];
};

__device__ __forceinline__ v8f wmma_bf16(v16bf a, v16bf b, v8f c) {
    return __builtin_amdgcn_wmma_f32_16x16x32_bf16(
        /*neg_a=*/false, a, /*neg_b=*/false, b,
        /*c_mod=*/(short)0, c, /*reuse_a=*/false, /*reuse_b=*/false);
}

// gfx1250 async copy: LDS[ldsOff] = MEM[gaddr], 16 bytes, tracked by ASYNCcnt.
__device__ __forceinline__ void async_copy_b128(unsigned int ldsOff, const void* gaddr) {
    asm volatile("global_load_async_to_lds_b128 %0, %1, off"
                 :: "v"(ldsOff), "v"((unsigned long long)(size_t)gaddr)
                 : "memory");
}

// ---------------------------------------------------------------------------
// Pack / convert kernels
// ---------------------------------------------------------------------------

// qin[m][0:512]=x[m], qin[m][512:1024]=y[m], converted to bf16
__global__ void concat_xy_bf16(const float* __restrict__ x,
                               const float* __restrict__ y,
                               __bf16* __restrict__ qin) {
    size_t i = (size_t)blockIdx.x * blockDim.x + threadIdx.x;   // over MTOT*FF
    size_t m = i >> 10;          // / 1024
    int    j = (int)(i & 1023);
    float  v = (j < QSZ) ? x[m * QSZ + j] : y[m * QSZ + (j - QSZ)];
    qin[i] = (__bf16)v;
}

__global__ void f32_to_bf16(const float* __restrict__ src,
                            __bf16* __restrict__ dst) {
    size_t i = (size_t)blockIdx.x * blockDim.x + threadIdx.x;
    dst[i] = (__bf16)src[i];
}

// WT[n][k] = (bf16) W[k][n], both 1024x1024
__global__ void w_transpose_bf16(const float* __restrict__ W,
                                 __bf16* __restrict__ WT) {
    size_t i = (size_t)blockIdx.x * blockDim.x + threadIdx.x;   // over 1024*1024
    int k = (int)(i & 1023);
    int n = (int)(i >> 10);
    WT[i] = (__bf16)W[(size_t)k * FF + n];
}

// v: [B,S,H,DH] bf16  ->  vT: [B,H,DH,S] bf16
__global__ void v_transpose_bf16(const __bf16* __restrict__ v,
                                 __bf16* __restrict__ vT) {
    size_t i = (size_t)blockIdx.x * blockDim.x + threadIdx.x;   // over B*H*DH*S
    int    s = (int)(i % SS);
    size_t t = i / SS;
    int    d = (int)(t % DHH); t /= DHH;
    int    h = (int)(t % HH);
    int    b = (int)(t / HH);
    vT[i] = v[(((size_t)b * SS + s) * HH + h) * DHH + d];
}

// ---------------------------------------------------------------------------
// bf16 WMMA GEMM:  C[M,N] = A[M,K] * B[K,N], with B given as Bt[N,K].
// Wave tile: 32(M) x 64(N). Block = 256 threads = 8 waves stacked in M
// -> block tile 256(M) x 64(N). grid = (N/64, M/256).
// The 64x32 B tile (shared by all 8 waves) is staged into LDS with
// double-buffered gfx1250 async copies (ASYNCcnt pipeline).
// ---------------------------------------------------------------------------
template <typename OutT>
__global__ __launch_bounds__(256)
void gemm_bf16_wmma(const __bf16* __restrict__ A,
                    const __bf16* __restrict__ Bt,
                    OutT* __restrict__ C,
                    int M, int N, int K) {
    __shared__ __align__(32) __bf16 bstage[2][64 * 32];  // 2 x 4 KiB B tiles

    const int tid   = threadIdx.x;
    const int lane  = tid & 31;
    const int wave  = tid >> 5;
    const int lrow  = lane & 15;
    const int lhalf = lane >> 4;

    const int n0 = blockIdx.x * 64;
    const int m0 = blockIdx.y * 256 + wave * 32;

    // Async-copy work split: thread t fetches 16B = Bt[n0 + t/4][ (t%4)*8 .. +7 ]
    const int bn = tid >> 2;
    const int bk = (tid & 3) * 8;
    const __bf16*      bsrc    = Bt + (size_t)(n0 + bn) * K + bk;
    const unsigned int ldsoff0 = (unsigned int)(size_t)&bstage[0][bn * 32 + bk];
    const unsigned int ldsoff1 = (unsigned int)(size_t)&bstage[1][bn * 32 + bk];

    v8f acc[2][4] = {};

    const __bf16* a0 = A + (size_t)(m0 + lrow) * K + lhalf * 16;
    const __bf16* a1 = A + (size_t)(m0 + 16 + lrow) * K + lhalf * 16;

    // prologue: stage K-tile 0 into buffer 0
    async_copy_b128(ldsoff0, bsrc);

    for (int k = 0; k < K; k += 32) {
        const int  cur  = (k >> 5) & 1;
        const bool more = (k + 32) < K;
        if (more) {
            // overlap next tile's copy with this tile's math
            async_copy_b128(cur ? ldsoff0 : ldsoff1, bsrc + k + 32);
            asm volatile("s_wait_asynccnt 0x1" ::: "memory");
        } else {
            asm volatile("s_wait_asynccnt 0x0" ::: "memory");
        }
        __syncthreads();   // make the staged tile visible block-wide

        FragA fa0, fa1;
        fa0.h[0] = *(const v8bf*)(a0 + k);
        fa0.h[1] = *(const v8bf*)(a0 + k + 8);
        fa1.h[0] = *(const v8bf*)(a1 + k);
        fa1.h[1] = *(const v8bf*)(a1 + k + 8);

        const __bf16* bbase = &bstage[cur][lrow * 32 + lhalf * 16];
#pragma unroll
        for (int nt = 0; nt < 4; ++nt) {
            v16bf b = *(const v16bf*)(bbase + nt * 16 * 32);
            acc[0][nt] = wmma_bf16(fa0.v, b, acc[0][nt]);
            acc[1][nt] = wmma_bf16(fa1.v, b, acc[1][nt]);
        }
        __syncthreads();   // all waves done reading before buffer is re-issued
    }

#pragma unroll
    for (int mi = 0; mi < 2; ++mi)
#pragma unroll
        for (int nt = 0; nt < 4; ++nt)
#pragma unroll
            for (int r = 0; r < 8; ++r) {
                int row = m0 + mi * 16 + r + lhalf * 8;
                int col = n0 + nt * 16 + lrow;
                C[(size_t)row * N + col] = (OutT)acc[mi][nt][r];
            }
}

// ---------------------------------------------------------------------------
// Flash attention, bf16 WMMA, online softmax.
// One wave = 16 query rows of one (b,h). Block = 8 waves = 128 query rows.
// grid = (S/128, H, B).
// q,k: [B,S,H,DH] bf16 (row stride FF).  vT: [B,H,DH,S] bf16.
// ao (output): [B,S,H,DH] bf16.
// ---------------------------------------------------------------------------
__global__ __launch_bounds__(256)
void flash_attn_bf16(const __bf16* __restrict__ qb,
                     const __bf16* __restrict__ kb,
                     const __bf16* __restrict__ vT,
                     __bf16* __restrict__ ao) {
    __shared__ __align__(32) __bf16 stage[8][16 * 32];  // per-wave P staging

    const int lane  = threadIdx.x & 31;
    const int wave  = threadIdx.x >> 5;
    const int lrow  = lane & 15;
    const int lhalf = lane >> 4;

    const int h  = blockIdx.y;
    const int b  = blockIdx.z;
    const int q0 = blockIdx.x * 128 + wave * 16;

    // Load the 16x64 Q tile as two A fragments (K = 0..31, 32..63)
    const __bf16* qp = qb + ((size_t)(b * SS + q0 + lrow)) * FF + h * DHH + lhalf * 16;
    FragA qa0, qa1;
    qa0.h[0] = *(const v8bf*)(qp);
    qa0.h[1] = *(const v8bf*)(qp + 8);
    qa1.h[0] = *(const v8bf*)(qp + 32);
    qa1.h[1] = *(const v8bf*)(qp + 40);

    v8f   acc[4] = {};
    float m_run[8], l_run[8];
#pragma unroll
    for (int r = 0; r < 8; ++r) { m_run[r] = -1e30f; l_run[r] = 0.0f; }

    const float scale = 0.125f;   // 1/sqrt(64)

    for (int kt = 0; kt < SS; kt += 32) {
        // ---- scores: two 16x16 key tiles, K-dim = DH = 64 (2 WMMA steps each)
        const __bf16* kp0 = kb + ((size_t)(b * SS + kt + lrow)) * FF + h * DHH + lhalf * 16;
        const __bf16* kp1 = kp0 + (size_t)16 * FF;
        v16bf b00 = *(const v16bf*)(kp0);
        v16bf b01 = *(const v16bf*)(kp0 + 32);
        v16bf b10 = *(const v16bf*)(kp1);
        v16bf b11 = *(const v16bf*)(kp1 + 32);

        v8f s0 = {}, s1 = {};
        s0 = wmma_bf16(qa0.v, b00, s0);
        s0 = wmma_bf16(qa1.v, b01, s0);
        s1 = wmma_bf16(qa0.v, b10, s1);
        s1 = wmma_bf16(qa1.v, b11, s1);

        // ---- online softmax per row (rows 0-7 in lanes 0-15, rows 8-15 in 16-31)
#pragma unroll
        for (int r = 0; r < 8; ++r) {
            float v0 = s0[r] * scale;
            float v1 = s1[r] * scale;

            float mx = fmaxf(v0, v1);
            mx = fmaxf(mx, __shfl_xor(mx, 1, 16));
            mx = fmaxf(mx, __shfl_xor(mx, 2, 16));
            mx = fmaxf(mx, __shfl_xor(mx, 4, 16));
            mx = fmaxf(mx, __shfl_xor(mx, 8, 16));

            float mt = fmaxf(m_run[r], mx);
            float p0 = __expf(v0 - mt);
            float p1 = __expf(v1 - mt);

            float ps = p0 + p1;
            ps += __shfl_xor(ps, 1, 16);
            ps += __shfl_xor(ps, 2, 16);
            ps += __shfl_xor(ps, 4, 16);
            ps += __shfl_xor(ps, 8, 16);

            float sc = __expf(m_run[r] - mt);
            l_run[r] = l_run[r] * sc + ps;
            m_run[r] = mt;
#pragma unroll
            for (int nt = 0; nt < 4; ++nt) acc[nt][r] *= sc;

            // stage P (C-layout -> row-major 16x32 tile in LDS)
            int row = r + lhalf * 8;
            stage[wave][row * 32 + lrow]      = (__bf16)p0;
            stage[wave][row * 32 + 16 + lrow] = (__bf16)p1;
        }

        // Wave-local LDS RAW: wait for our own DS stores before A-frag reload.
        asm volatile("s_wait_dscnt 0x0" ::: "memory");

        // ---- reload P as an A fragment (16x32, K = key-in-tile)
        FragA pa;
        const __bf16* sp = &stage[wave][lrow * 32 + lhalf * 16];
        pa.h[0] = *(const v8bf*)(sp);
        pa.h[1] = *(const v8bf*)(sp + 8);

        // ---- acc += P x V   (B operand from vT, contiguous in key dim)
        const __bf16* vp = vT + ((size_t)(b * HH + h) * DHH + lrow) * SS + kt + lhalf * 16;
#pragma unroll
        for (int nt = 0; nt < 4; ++nt) {
            v16bf bv = *(const v16bf*)(vp + (size_t)nt * 16 * SS);
            acc[nt] = wmma_bf16(pa.v, bv, acc[nt]);
        }
    }

    // ---- epilogue: normalize and store [B,S,H,DH] bf16
#pragma unroll
    for (int nt = 0; nt < 4; ++nt)
#pragma unroll
        for (int r = 0; r < 8; ++r) {
            int row = q0 + r + lhalf * 8;
            int col = h * DHH + nt * 16 + lrow;
            ao[(size_t)(b * SS + row) * FF + col] = (__bf16)(acc[nt][r] / l_run[r]);
        }
}

// ---------------------------------------------------------------------------
// Host launcher
// ---------------------------------------------------------------------------
extern "C" void kernel_launch(void* const* d_in, const int* in_sizes, int n_in,
                              void* d_out, int out_size, void* d_ws, size_t ws_size,
                              hipStream_t stream) {
    const float* x  = (const float*)d_in[0];
    const float* y  = (const float*)d_in[1];
    const float* z  = (const float*)d_in[2];
    const float* Wq = (const float*)d_in[3];
    const float* Wk = (const float*)d_in[4];
    const float* Wv = (const float*)d_in[5];
    const float* Wo = (const float*)d_in[6];
    float* out = (float*)d_out;

    char*  ws  = (char*)d_ws;
    size_t off = 0;
    auto take = [&](size_t bytes) {
        char* p = ws + off;
        off += (bytes + 255) & ~(size_t)255;
        return p;
    };

    const size_t actBytes = (size_t)MTOT * FF * sizeof(__bf16);   // 16 MiB
    const size_t wBytes   = (size_t)FF * FF * sizeof(__bf16);     // 2 MiB

    __bf16* qin = (__bf16*)take(actBytes);
    __bf16* zb  = (__bf16*)take(actBytes);
    __bf16* WqT = (__bf16*)take(wBytes);
    __bf16* WkT = (__bf16*)take(wBytes);
    __bf16* WvT = (__bf16*)take(wBytes);
    __bf16* WoT = (__bf16*)take(wBytes);
    __bf16* qb  = (__bf16*)take(actBytes);
    __bf16* kb  = (__bf16*)take(actBytes);
    __bf16* vb  = (__bf16*)take(actBytes);
    __bf16* vT  = (__bf16*)take(actBytes);
    __bf16* ao  = (__bf16*)take(actBytes);
    (void)ws_size; (void)in_sizes; (void)n_in; (void)out_size;

    const int  T  = 256;
    const int  gAct = (int)(((size_t)MTOT * FF) / T);   // 32768
    const int  gW   = (int)(((size_t)FF * FF) / T);     // 4096

    // 1. pack inputs / weights into bf16 WMMA-friendly layouts
    concat_xy_bf16<<<gAct, T, 0, stream>>>(x, y, qin);
    f32_to_bf16<<<gAct, T, 0, stream>>>(z, zb);
    w_transpose_bf16<<<gW, T, 0, stream>>>(Wq, WqT);
    w_transpose_bf16<<<gW, T, 0, stream>>>(Wk, WkT);
    w_transpose_bf16<<<gW, T, 0, stream>>>(Wv, WvT);
    w_transpose_bf16<<<gW, T, 0, stream>>>(Wo, WoT);

    // 2. projections (bf16 WMMA GEMMs, async-LDS staged B)
    dim3 gemmGrid(FF / 64, MTOT / 256);
    gemm_bf16_wmma<__bf16><<<gemmGrid, T, 0, stream>>>(qin, WqT, qb, MTOT, FF, FF);
    gemm_bf16_wmma<__bf16><<<gemmGrid, T, 0, stream>>>(zb,  WkT, kb, MTOT, FF, FF);
    gemm_bf16_wmma<__bf16><<<gemmGrid, T, 0, stream>>>(zb,  WvT, vb, MTOT, FF, FF);

    // 3. transpose V for the PV B-operand
    v_transpose_bf16<<<gAct, T, 0, stream>>>(vb, vT);

    // 4. flash attention
    dim3 attnGrid(SS / 128, HH, BB);
    flash_attn_bf16<<<attnGrid, T, 0, stream>>>(qb, kb, vT, ao);

    // 5. output projection -> f32
    gemm_bf16_wmma<float><<<gemmGrid, T, 0, stream>>>(ao, WoT, out, MTOT, FF, FF);
}
